// GlobalAttention_69458211111344
// MI455X (gfx1250) — matrix operand order repaired and checked by reading, not compiled, not run on hardware
//
#include <hip/hip_runtime.h>
#include <hip/hip_bf16.h>
#include <cstdint>
#include <cstddef>

// ---------------------------------------------------------------------------
// Problem constants (from the reference):
//   B=4, T=512, S=512, V=8, M=2, K=4, F=1, Hqk=32, Hv=32, C=256
// ---------------------------------------------------------------------------
constexpr int Bn  = 4;
constexpr int Tn  = 512;
constexpr int Sn  = 512;
constexpr int Vn  = 8;
constexpr int Mn  = 2;
constexpr int Kn  = 4;
constexpr int Hqk = 32;
constexpr int Hv  = 32;
constexpr int Cn  = 256;

constexpr int NR = Bn * Tn * Vn;   // 16384 rows of X / Q / att
constexpr int QC = Mn * Kn * Hqk;  // 256 = q cols and att cols
constexpr int KC = Kn * Hqk;       // 128 = k/v cols

typedef __attribute__((ext_vector_type(16))) _Float16 v16h;
typedef __attribute__((ext_vector_type(8)))  float    v8f;

union Frag { v16h h; uint32_t u[8]; };

// 16-bit A-matrix 16x32 fragment: lane holds row (lane&15); element pair base:
//   k = (vg>=4 ? 16:0) | (lane>=16 ? 8:0) | (vg&3)*2   (+half)
__device__ __forceinline__ int kpatA(int vg, int lane) {
  return ((vg >> 2) << 4) | (((lane >> 4) & 1) << 3) | ((vg & 3) << 1);
}
// 16-bit B-matrix 32x16 fragment: lane holds col (lane&15); element pair base:
//   k = (vg>=4 ? 16:0) | (lane>=16 ? 8:0) | (vg&3)*2   (+half)
__device__ __forceinline__ int kpatB(int vg, int lane) {
  return ((vg >> 2) << 4) | (((lane >> 4) & 1) << 3) | ((vg & 3) << 1);
}

// 16-lane xor-butterfly max. Prefer v_permlane16_b32 (pure VALU) over
// ds_bpermute-based __shfl_xor to keep LDS free for the WMMA staging.
__device__ __forceinline__ float rowmax16(float x) {
#if __has_builtin(__builtin_amdgcn_permlane16)
  // selector tables (per-lane 4-bit src within each 16-lane row), low nibble
  // = lane 0 / lane 8 of the row half.
  const uint32_t s0[4] = {0x67452301u, 0x54761032u, 0x32107654u, 0xFEDCBA98u};
  const uint32_t s1[4] = {0xEFCDAB89u, 0xDCFE98BAu, 0xBA98FEDCu, 0x76543210u};
#pragma unroll
  for (int i = 0; i < 4; ++i) {
    uint32_t u = __builtin_bit_cast(uint32_t, x);
    uint32_t p = __builtin_amdgcn_permlane16(u, u, s0[i], s1[i], false, false);
    x = fmaxf(x, __builtin_bit_cast(float, p));
  }
#else
#pragma unroll
  for (int d = 1; d < 16; d <<= 1) x = fmaxf(x, __shfl_xor(x, d, 32));
#endif
  return x;
}

// ---------------------------------------------------------------------------
// f32 -> f16 conversion, two elements per thread, packed 32-bit stores.
// All array lengths here are even.
// ---------------------------------------------------------------------------
__global__ void f32_to_f16_kernel(const float* __restrict__ in,
                                  uint32_t* __restrict__ out, int n2) {
  int i = blockIdx.x * blockDim.x + threadIdx.x;
  int stride = gridDim.x * blockDim.x;
  for (; i < n2; i += stride) {
    float a = in[2 * i], b = in[2 * i + 1];
    union { _Float16 h[2]; uint32_t u; } p;
    p.h[0] = (_Float16)a;
    p.h[1] = (_Float16)b;
    out[i] = p.u;
  }
}

// ---------------------------------------------------------------------------
// Tiled f16 GEMM with f32 accumulation: C(MxN) = A(MxKd) * B(KdxN)
// A, B row-major f16. Block tile 128x64, BK=32, 8 waves (256 thr, wave32).
// Each wave owns a 32x32 sub-tile = 4 x v_wmma_f32_16x16x32_f16 per K step.
// Requires Md%128==0, Nd%64==0, Kd%32==0 (true for all uses here).
// ---------------------------------------------------------------------------
template <bool OUT_F32>
__global__ __launch_bounds__(256) void gemm16(const _Float16* __restrict__ A,
                                              const _Float16* __restrict__ Bm,
                                              void* __restrict__ Cout,
                                              int Md, int Nd, int Kd) {
  constexpr int BM = 128, BN = 64, BK = 32;
  __shared__ _Float16 As[BM * BK];    // 8 KB
  __shared__ _Float16 BsT[BN * BK];   // 4 KB, transposed: BsT[n][k]

  const int tid  = threadIdx.x;
  const int lane = tid & 31;
  const int wave = tid >> 5;
  const int wm   = wave & 3;   // 4 waves along M
  const int wn   = wave >> 2;  // 2 waves along N
  const int m0   = blockIdx.x * BM;
  const int n0   = blockIdx.y * BN;

  v8f acc[2][2] = {};

  for (int k0 = 0; k0 < Kd; k0 += BK) {
    // Stage A tile (128x32): 256 threads x 8 iters x ushort2
#pragma unroll
    for (int i = 0; i < 8; ++i) {
      int idx = tid + i * 256;
      int r = idx >> 4, cp = (idx & 15) << 1;
      *(uint32_t*)&As[r * BK + cp] =
          *(const uint32_t*)&A[(size_t)(m0 + r) * Kd + k0 + cp];
    }
    // Stage B tile (32x64) transposed into BsT[n][k]
#pragma unroll
    for (int i = 0; i < 8; ++i) {
      int idx = tid + i * 256;
      int kk = idx >> 6, nn = idx & 63;
      BsT[nn * BK + kk] = Bm[(size_t)(k0 + kk) * Nd + n0 + nn];
    }
    // Prefetch next A tile rows (lowers to global_prefetch_b8)
    if (k0 + BK < Kd)
      __builtin_prefetch(&A[(size_t)(m0 + (tid >> 1)) * Kd + k0 + BK], 0, 0);
    __syncthreads();

    Frag a[2], b[2];
#pragma unroll
    for (int i = 0; i < 2; ++i) {
      int rb = (wm * 32 + i * 16 + (lane & 15)) * BK;
#pragma unroll
      for (int vg = 0; vg < 8; ++vg)
        a[i].u[vg] = *(const uint32_t*)&As[rb + kpatA(vg, lane)];
    }
#pragma unroll
    for (int j = 0; j < 2; ++j) {
      int nb = (wn * 32 + j * 16 + (lane & 15)) * BK;
#pragma unroll
      for (int vg = 0; vg < 8; ++vg)
        b[j].u[vg] = *(const uint32_t*)&BsT[nb + kpatB(vg, lane)];
    }
#pragma unroll
    for (int i = 0; i < 2; ++i)
#pragma unroll
      for (int j = 0; j < 2; ++j)
        acc[i][j] = __builtin_amdgcn_wmma_f32_16x16x32_f16(
            false, a[i].h, false, b[j].h, (short)0, acc[i][j], false, false);
    __syncthreads();
  }

  // Epilogue: C/D layout — VGPR r: row = r + (lane>=16 ? 8:0), col = lane&15
#pragma unroll
  for (int i = 0; i < 2; ++i)
#pragma unroll
    for (int j = 0; j < 2; ++j)
#pragma unroll
      for (int r = 0; r < 8; ++r) {
        int row = m0 + wm * 32 + i * 16 + r + ((lane >> 4) << 3);
        int col = n0 + wn * 32 + j * 16 + (lane & 15);
        if (OUT_F32)
          ((float*)Cout)[(size_t)row * Nd + col] = acc[i][j][r];
        else
          ((_Float16*)Cout)[(size_t)row * Nd + col] = (_Float16)acc[i][j][r];
      }
}

// ---------------------------------------------------------------------------
// Fused masked attention (flash-attention style), head dim 32.
// grid.x = B*V*M*K heads (256); grid.y = T/64; block = 128 threads (4 waves).
// Each wave handles 16 query rows; loops S in steps of 32:
//   logits = Q(16x32) * K^T(32x32)   -> 2 WMMA
//   row max: permlane16 butterfly (VALU, no LDS)
//   P staged through per-wave LDS -> A fragment
//   row sums of P via WMMA against all-ones B fragment -> 1 WMMA
//   O += P(16x32) * V(32x32)         -> 2 WMMA
// ---------------------------------------------------------------------------
__global__ __launch_bounds__(128) void attn_kernel(
    const _Float16* __restrict__ qh, const _Float16* __restrict__ kh,
    const _Float16* __restrict__ vh, const unsigned char* __restrict__ mask,
    _Float16* __restrict__ atth, float scale) {
  __shared__ _Float16 Plds[4][16 * 32];  // per-wave P staging, 4 KB total

  const int lane = threadIdx.x & 31;
  const int wave = threadIdx.x >> 5;
  const int ln = lane & 15;
  const int hi = (lane >> 4) & 1;

  const int head = blockIdx.x;               // (b, v, m, k) flattened
  const int b   = head >> 6;                 // / (V*M*K = 64)
  const int rem = head & 63;
  const int v   = rem >> 3;                  // / (M*K = 8)
  const int mk  = rem & 7;                   // m*K + k
  const int kk  = mk & 3;                    // k (kv head)

  const int tbase = blockIdx.y * 64 + wave * 16;

  // Q fragment: A-layout, loaded straight from global (row-major, pairs contig)
  Frag aq;
  {
    const _Float16* qrow =
        qh + ((size_t)(b * Tn + tbase + ln) * Vn + v) * QC + mk * Hqk;
#pragma unroll
    for (int vg = 0; vg < 8; ++vg)
      aq.u[vg] = *(const uint32_t*)&qrow[kpatA(vg, lane)];
  }

  // All-ones B fragment (f16 1.0 pairs) for WMMA-based row sums of P.
  Frag ones;
#pragma unroll
  for (int vg = 0; vg < 8; ++vg) ones.u[vg] = 0x3C003C00u;

  float rm[8], rl[8], al[8];
#pragma unroll
  for (int r = 0; r < 8; ++r) { rm[r] = -__builtin_inff(); rl[r] = 0.f; }
  v8f o0 = {}, o1 = {};
  const v8f zacc = {};
  const unsigned short* vhu = (const unsigned short*)vh;

  for (int s0 = 0; s0 < Sn; s0 += 32) {
    // K^T B fragments: element (dim, s) read from kh[s][dim] (dim pairs contig)
    Frag bk[2];
#pragma unroll
    for (int j = 0; j < 2; ++j) {
      int s = s0 + j * 16 + ln;
      const _Float16* krow =
          kh + ((size_t)(b * Sn + s) * Vn + v) * KC + kk * Hqk;
#pragma unroll
      for (int vg = 0; vg < 8; ++vg)
        bk[j].u[vg] = *(const uint32_t*)&krow[kpatB(vg, lane)];
    }
    v8f l0 = __builtin_amdgcn_wmma_f32_16x16x32_f16(false, aq.h, false, bk[0].h,
                                                    (short)0, zacc, false, false);
    v8f l1 = __builtin_amdgcn_wmma_f32_16x16x32_f16(false, aq.h, false, bk[1].h,
                                                    (short)0, zacc, false, false);

    // mask + online softmax; each VGPR r holds row (r + hi*8), col (lane&15)
#pragma unroll
    for (int r = 0; r < 8; ++r) {
      int t = tbase + r + hi * 8;
      const unsigned char* mrow = mask + ((size_t)b * Tn + t) * Sn + s0;
      float p0 = l0[r] * scale;
      float p1 = l1[r] * scale;
      if (!mrow[ln])      p0 = -__builtin_inff();
      if (!mrow[16 + ln]) p1 = -__builtin_inff();
      float mx = rowmax16(fmaxf(p0, p1));
      float nm = fmaxf(rm[r], mx);
      al[r] = __expf(rm[r] - nm);  // first tile: exp(-inf)=0; later: finite
      rm[r] = nm;
      p0 = __expf(p0 - nm);
      p1 = __expf(p1 - nm);
      o0[r] *= al[r];
      o1[r] *= al[r];
      int prow = r + hi * 8;
      Plds[wave][prow * 32 + ln]      = (_Float16)p0;
      Plds[wave][prow * 32 + 16 + ln] = (_Float16)p1;
    }
    __syncthreads();  // uniform trip count across all 4 waves

    // P back as A-fragment
    Frag ap;
#pragma unroll
    for (int vg = 0; vg < 8; ++vg)
      ap.u[vg] = *(const uint32_t*)&Plds[wave][ln * 32 + kpatA(vg, lane)];

    // Row sums of P via matrix core: P(16x32) x J(32x16) -> every column of
    // the C tile holds the row sum, i.e. each lane gets its own row's sum.
    v8f srow = __builtin_amdgcn_wmma_f32_16x16x32_f16(
        false, ap.h, false, ones.h, (short)0, zacc, false, false);
#pragma unroll
    for (int r = 0; r < 8; ++r) rl[r] = rl[r] * al[r] + srow[r];

    // V B-fragments: element (s, dv) from vh[s][dv]; s pairs strided -> 2xu16
    Frag bv[2];
#pragma unroll
    for (int jv = 0; jv < 2; ++jv) {
      int dv = jv * 16 + ln;
#pragma unroll
      for (int vg = 0; vg < 8; ++vg) {
        int sa = s0 + ((vg >> 2) << 4) + hi * 8 + ((vg & 3) << 1);
        uint32_t lo = vhu[((size_t)(b * Sn + sa) * Vn + v) * KC + kk * Hv + dv];
        uint32_t h2 = vhu[((size_t)(b * Sn + sa + 1) * Vn + v) * KC + kk * Hv + dv];
        bv[jv].u[vg] = lo | (h2 << 16);
      }
    }
    o0 = __builtin_amdgcn_wmma_f32_16x16x32_f16(false, ap.h, false, bv[0].h,
                                                (short)0, o0, false, false);
    o1 = __builtin_amdgcn_wmma_f32_16x16x32_f16(false, ap.h, false, bv[1].h,
                                                (short)0, o1, false, false);
    __syncthreads();
  }

  // Normalize and store att rows (f16) for the output projection GEMM
#pragma unroll
  for (int r = 0; r < 8; ++r) {
    float inv = 1.f / rl[r];
    int t = tbase + r + hi * 8;
    _Float16* orow = atth + ((size_t)(b * Tn + t) * Vn + v) * QC + mk * Hv;
    orow[ln]      = (_Float16)(o0[r] * inv);
    orow[16 + ln] = (_Float16)(o1[r] * inv);
  }
}

// ---------------------------------------------------------------------------
// Launch: convert -> QKV GEMMs -> fused attention -> output GEMM (f32 out)
// Workspace (f16): xh | Wqh | Wkh | Wvh | Wouth | q | k | v | att  (~34 MB)
// ---------------------------------------------------------------------------
extern "C" void kernel_launch(void* const* d_in, const int* in_sizes, int n_in,
                              void* d_out, int out_size, void* d_ws,
                              size_t ws_size, hipStream_t stream) {
  const float* x    = (const float*)d_in[0];
  const float* Wq   = (const float*)d_in[1];
  const float* Wk   = (const float*)d_in[2];
  const float* Wv   = (const float*)d_in[3];
  const float* Wout = (const float*)d_in[4];
  const unsigned char* mask = (const unsigned char*)d_in[5];  // jnp bool = 1B
  float* out = (float*)d_out;

  _Float16* ws = (_Float16*)d_ws;
  size_t off = 0;
  _Float16* xh    = ws + off; off += (size_t)NR * Cn;
  _Float16* Wqh   = ws + off; off += (size_t)Cn * QC;
  _Float16* Wkh   = ws + off; off += (size_t)Cn * KC;
  _Float16* Wvh   = ws + off; off += (size_t)Cn * KC;
  _Float16* Wouth = ws + off; off += (size_t)QC * Cn;
  _Float16* qhb   = ws + off; off += (size_t)NR * QC;
  _Float16* khb   = ws + off; off += (size_t)NR * KC;
  _Float16* vhb   = ws + off; off += (size_t)NR * KC;
  _Float16* atth  = ws + off; off += (size_t)NR * QC;

  auto conv = [&](const float* a, _Float16* b, int n) {
    int n2 = n / 2;
    int blocks = (n2 + 255) / 256;
    if (blocks > 2048) blocks = 2048;
    f32_to_f16_kernel<<<dim3(blocks), dim3(256), 0, stream>>>(
        a, (uint32_t*)b, n2);
  };
  conv(x,    xh,    NR * Cn);
  conv(Wq,   Wqh,   Cn * QC);
  conv(Wk,   Wkh,   Cn * KC);
  conv(Wv,   Wvh,   Cn * KC);
  conv(Wout, Wouth, QC * Cn);

  // QKV projections
  gemm16<false><<<dim3(NR / 128, QC / 64), 256, 0, stream>>>(xh, Wqh, qhb, NR, QC, Cn);
  gemm16<false><<<dim3(NR / 128, KC / 64), 256, 0, stream>>>(xh, Wkh, khb, NR, KC, Cn);
  gemm16<false><<<dim3(NR / 128, KC / 64), 256, 0, stream>>>(xh, Wvh, vhb, NR, KC, Cn);

  // Fused masked attention over 256 heads, scale = 1/sqrt(F*Hqk)
  const float scale = 0.1767766952966369f;  // 1/sqrt(32)
  attn_kernel<<<dim3(Bn * Vn * Mn * Kn, Tn / 64), 128, 0, stream>>>(
      qhb, khb, vhb, mask, atth, scale);

  // Output projection, f32 result straight into d_out (b,t,v,c)
  gemm16<true><<<dim3(NR / 128, Cn / 64), 256, 0, stream>>>(atth, Wouth, out, NR, Cn, Cn);
}
